// GPT2Attention_27436251086985
// MI455X (gfx1250) — compile-verified, matrix operand order checked by reading, not run on hardware
//
#include <hip/hip_runtime.h>
#include <hip/hip_bf16.h>

#define BATCH  4
#define SEQ    2048
#define DMODEL 1024
#define NH     16
#define HD     64
#define QKSCALE 0.125f          // 1/sqrt(64)
#define MASK_BIAS -10000.0f

typedef __attribute__((ext_vector_type(16))) __bf16        v16bf;
typedef __attribute__((ext_vector_type(8)))  float         v8f;
typedef __attribute__((ext_vector_type(4)))  unsigned int  v4u;
typedef __attribute__((ext_vector_type(4)))  unsigned int  u32x4;
typedef __attribute__((ext_vector_type(8)))  int           i32x8;
typedef __attribute__((ext_vector_type(4)))  int           i32x4;

#if __has_builtin(__builtin_amdgcn_tensor_load_to_lds) && __has_builtin(__builtin_amdgcn_s_wait_tensorcnt)
#define HAVE_TDM 1
#else
#define HAVE_TDM 0
#endif

union FragU {            // 16 bf16 elements = A/B operand of v_wmma_f32_16x16x32_bf16
    v4u   u[2];
    v16bf v;
};

__device__ __forceinline__ unsigned short f2bf(float x) {
    unsigned int u = __builtin_bit_cast(unsigned int, x);
    u += 0x7FFFu + ((u >> 16) & 1u);           // round-to-nearest-even
    return (unsigned short)(u >> 16);
}

__device__ __forceinline__ v8f wmma_bf16(const FragU& a, const FragU& b, v8f c) {
    return __builtin_amdgcn_wmma_f32_16x16x32_bf16(
        false, a.v, false, b.v, (short)0, c, false, false);
}

#if HAVE_TDM
// TDM: DMA a 2D bf16 tile (tile_w x tile_h elements, row-major, row stride
// `row_stride` elements) from global into LDS at byte offset lds_off, adding
// pad_amount DWORDs of LDS padding after each 2^pad_interval 8-byte groups.
// D# bit layout per CDNA5 ISA ch.7/8 (group0: flags/lds/global/type=2,
// group1: data_size, pad cfg, tensor+tile dims, dim0 stride).
__device__ __forceinline__ void tdm_load_2d_bf16(unsigned lds_off, const void* gaddr,
                                                 int tile_w, int tile_h,
                                                 int tensor_w, int tensor_h,
                                                 int row_stride,
                                                 int pad_interval_code, int pad_amount_code)
{
    unsigned long long ga = (unsigned long long)(uintptr_t)gaddr;
    u32x4 g0;
    g0[0] = 1u;                                        // count=1, user descriptor
    g0[1] = lds_off;                                   // lds_addr
    g0[2] = (unsigned)(ga & 0xFFFFFFFFu);              // global_addr[31:0]
    g0[3] = (unsigned)((ga >> 32) & 0x01FFFFFFu) | (2u << 30);  // addr[56:32] | type=2
    unsigned w0 = (1u << 16);                          // data_size = 1 (2 bytes)
    w0 |= (1u << 20)                                   // pad_enable
        | ((unsigned)pad_interval_code << 22)
        | ((unsigned)pad_amount_code << 25);
    i32x8 g1;
    g1[0] = (int)w0;
    g1[1] = (int)((unsigned)(tensor_w & 0xFFFF) << 16);                  // tensor_dim0[15:0]
    g1[2] = (int)(((unsigned)tensor_w >> 16) |
                  ((unsigned)(tensor_h & 0xFFFF) << 16));                // dim0[31:16] | dim1[15:0]
    g1[3] = (int)(((unsigned)tensor_h >> 16) |
                  ((unsigned)(tile_w & 0xFFFF) << 16));                  // dim1[31:16] | tile_dim0
    g1[4] = (int)((unsigned)(tile_h & 0xFFFF));                          // tile_dim1 (tile_dim2=0)
    g1[5] = (int)row_stride;                                             // tensor_dim0_stride[31:0]
    g1[6] = 0;
    g1[7] = 0;
    i32x4 z4 = {0, 0, 0, 0};
#if defined(__clang_major__) && (__clang_major__ >= 23)
    i32x8 z8 = {0, 0, 0, 0, 0, 0, 0, 0};
    __builtin_amdgcn_tensor_load_to_lds(g0, g1, z4, z4, z8, 0);
#else
    __builtin_amdgcn_tensor_load_to_lds(g0, g1, z4, z4, 0);
#endif
}

__device__ __forceinline__ unsigned lds_offset_of(const void* p) {
    // generic pointers into the LDS aperture carry the LDS byte offset in [31:0]
    return (unsigned)(uintptr_t)p;
}
#endif // HAVE_TDM

// ---------------------------------------------------------------- cast kernel
__global__ __launch_bounds__(256)
void cast_f32_bf16(const float* __restrict__ src, unsigned short* __restrict__ dst, int n) {
    int i = blockIdx.x * blockDim.x + threadIdx.x;
    int stride = gridDim.x * blockDim.x;
    for (; i < n; i += stride) dst[i] = f2bf(src[i]);
}

// ---------------------------------------------------------------- WMMA GEMM
// C[M,N] = A[M,K](bf16) * B[K,N](bf16) + bias[N]
// MODE 0: store f32 row-major to Cf.
// MODE 1: scatter bf16 into Q/K/V buffers with [B,H,S,hd] layout (N = 3*DMODEL).
template<int MODE>
__global__ __launch_bounds__(256)
void gemm_bf16(const unsigned short* __restrict__ A,
               const unsigned short* __restrict__ Bm,
               const float* __restrict__ bias,
               float* __restrict__ Cf,
               unsigned short* __restrict__ Qo,
               unsigned short* __restrict__ Ko,
               unsigned short* __restrict__ Vo,
               int M, int N, int K)
{
    constexpr int BM = 128, BN = 128, BK = 32, LDS_LD = BK + 8;  // 40 elems = 80B rows
    __shared__ __align__(16) unsigned short As[BM * LDS_LD];
    __shared__ __align__(16) unsigned short Bs[BN * LDS_LD];     // stored transposed: [n][k]

    const int tid   = threadIdx.x;
    const int lane  = tid & 31;
    const int wv    = tid >> 5;
    const int wm0   = (wv & 3) * 32;      // wave row offset in tile
    const int wn0   = (wv >> 2) * 64;     // wave col offset in tile
    const int l16   = lane & 15;
    const int lhalf = lane >> 4;

    const int bm0 = blockIdx.y * BM;
    const int bn0 = blockIdx.x * BN;

    // staging coordinates
    const int ar = tid >> 1;              // A: row 0..127
    const int ac = (tid & 1) * 16;        // A: col 0 or 16
    const int bk = tid >> 3;              // B: k 0..31
    const int bn = (tid & 7) * 16;        // B: n 0..112

    v8f acc[2][4];
#pragma unroll
    for (int mf = 0; mf < 2; ++mf)
#pragma unroll
        for (int nf = 0; nf < 4; ++nf)
            acc[mf][nf] = (v8f){0.f,0.f,0.f,0.f,0.f,0.f,0.f,0.f};

    const int ksteps = K / BK;
    for (int kt = 0; kt < ksteps; ++kt) {
        const int k0 = kt * BK;
        __syncthreads();
#if HAVE_TDM
        // A tile: one TDM descriptor per workgroup (wave 0 issues; EXEC ignored).
        // rows of 32 elems (16 DWORDs -> interval code 3), pad 4 DWORDs (code 3)
        if (wv == 0) {
            tdm_load_2d_bf16(lds_offset_of(As), A + (size_t)bm0 * K + k0,
                             BK, BM, K, M, K, /*interval=*/3, /*pad=*/3);
        }
#else
        { // stage A tile (row-major, contiguous 32B per thread)
            const v4u* g = (const v4u*)(A + (size_t)(bm0 + ar) * K + k0 + ac);
            v4u x0 = g[0], x1 = g[1];
            v4u* s = (v4u*)(As + ar * LDS_LD + ac);
            s[0] = x0; s[1] = x1;
        }
#endif
        { // stage B tile transposed into [n][k]
            const v4u* g = (const v4u*)(Bm + (size_t)(k0 + bk) * N + bn0 + bn);
            v4u x0 = g[0], x1 = g[1];
            union { v4u v[2]; unsigned short s[16]; } tmp;
            tmp.v[0] = x0; tmp.v[1] = x1;
#pragma unroll
            for (int j = 0; j < 16; ++j)
                Bs[(bn + j) * LDS_LD + bk] = tmp.s[j];
        }
        if (kt + 1 < ksteps) {
            __builtin_prefetch((const void*)(A + (size_t)(bm0 + ar) * K + k0 + BK + ac), 0, 0);
            __builtin_prefetch((const void*)(Bm + (size_t)(k0 + BK + bk) * N + bn0 + bn), 0, 0);
        }
#if HAVE_TDM
        if (wv == 0) __builtin_amdgcn_s_wait_tensorcnt(0);
#endif
        __syncthreads();

        FragU af[2], bfr[4];
#pragma unroll
        for (int mf = 0; mf < 2; ++mf) {
            const unsigned short* p = As + (wm0 + mf * 16 + l16) * LDS_LD + lhalf * 8;
            af[mf].u[0] = *(const v4u*)p;
            af[mf].u[1] = *(const v4u*)(p + 16);
        }
#pragma unroll
        for (int nf = 0; nf < 4; ++nf) {
            const unsigned short* p = Bs + (wn0 + nf * 16 + l16) * LDS_LD + lhalf * 8;
            bfr[nf].u[0] = *(const v4u*)p;
            bfr[nf].u[1] = *(const v4u*)(p + 16);
        }
#pragma unroll
        for (int mf = 0; mf < 2; ++mf)
#pragma unroll
            for (int nf = 0; nf < 4; ++nf)
                acc[mf][nf] = wmma_bf16(af[mf], bfr[nf], acc[mf][nf]);
    }

    // epilogue: C layout — lane l16 = N index, VGPR r = row r (+8 for upper half-wave)
#pragma unroll
    for (int mf = 0; mf < 2; ++mf)
#pragma unroll
        for (int nf = 0; nf < 4; ++nf)
#pragma unroll
            for (int r = 0; r < 8; ++r) {
                const int gm = bm0 + wm0 + mf * 16 + lhalf * 8 + r;
                const int gn = bn0 + wn0 + nf * 16 + l16;
                const float v = acc[mf][nf][r] + bias[gn];
                if (MODE == 0) {
                    Cf[(size_t)gm * N + gn] = v;
                } else {
                    const int part = gn >> 10;   // 0:Q 1:K 2:V
                    const int d    = gn & 1023;
                    const int h    = d >> 6;
                    const int e    = d & 63;
                    const int b    = gm >> 11;   // / SEQ
                    const int s    = gm & 2047;
                    const size_t idx = ((size_t)((b * NH + h) * SEQ + s)) * HD + e;
                    const unsigned short bv = f2bf(v);
                    if (part == 0)      Qo[idx] = bv;
                    else if (part == 1) Ko[idx] = bv;
                    else                Vo[idx] = bv;
                }
            }
}

// ---------------------------------------------------------------- flash attention
// One workgroup per (b,h, 64-row q block); 4 waves, 16 q rows per wave.
__global__ __launch_bounds__(128)
void flash_attn(const unsigned short* __restrict__ Qm,
                const unsigned short* __restrict__ Km,
                const unsigned short* __restrict__ Vm,
                unsigned short* __restrict__ Om)
{
    constexpr int QLD = HD + 8;   // 72 elems = 144B rows (16B aligned)
    constexpr int KLD = HD + 8;
    constexpr int VLD = 32 + 8;   // 40 elems = 80B rows
    constexpr int PLD = 32 + 8;
    __shared__ __align__(16) unsigned short Qs[64 * QLD];
    __shared__ __align__(16) unsigned short Ks[32 * KLD];   // [key][d]  (B-frag for Q*K^T)
    __shared__ __align__(16) unsigned short Vt[HD * VLD];   // [d][key]  (B-frag for P*V)
    __shared__ __align__(16) unsigned short Ps[64 * PLD];   // [qrow][key] per-wave P tiles

    const int tid   = threadIdx.x;
    const int lane  = tid & 31;
    const int w     = tid >> 5;
    const int l16   = lane & 15;
    const int lhalf = lane >> 4;

    const int nq   = SEQ / 64;               // 32 q blocks per (b,h)
    const int qblk = blockIdx.x % nq;
    const int bh   = blockIdx.x / nq;
    const size_t rowbase = (size_t)bh * SEQ;

#if HAVE_TDM
    // stage the 64x64 Q block via TDM (rows of 64 elems = 32 DW -> interval 4, pad 4 DW)
    if (w == 0) {
        tdm_load_2d_bf16(lds_offset_of(Qs), Qm + (rowbase + (size_t)qblk * 64) * HD,
                         HD, 64, HD, SEQ, HD, /*interval=*/4, /*pad=*/3);
        __builtin_amdgcn_s_wait_tensorcnt(0);
    }
#else
    { // stage the 64x64 Q block
        const int r = tid >> 1;
        const int c = (tid & 1) * 32;
        const v4u* g = (const v4u*)(Qm + (rowbase + qblk * 64 + r) * HD + c);
        v4u* s = (v4u*)(Qs + r * QLD + c);
        s[0] = g[0]; s[1] = g[1]; s[2] = g[2]; s[3] = g[3];
    }
#endif
    __syncthreads();

    FragU aq[2];                              // Q fragments stay resident (K=64 -> 2 k-steps)
#pragma unroll
    for (int ks = 0; ks < 2; ++ks) {
        const unsigned short* p = Qs + (w * 16 + l16) * QLD + ks * 32 + lhalf * 8;
        aq[ks].u[0] = *(const v4u*)p;
        aq[ks].u[1] = *(const v4u*)(p + 16);
    }

    v8f o[4];
#pragma unroll
    for (int nf = 0; nf < 4; ++nf) o[nf] = (v8f){0.f,0.f,0.f,0.f,0.f,0.f,0.f,0.f};
    float rm[8], rl[8];
#pragma unroll
    for (int r = 0; r < 8; ++r) { rm[r] = -1e30f; rl[r] = 0.f; }

    const int q0  = qblk * 64 + w * 16;
    const int nkb = (qblk * 64 + 64) / 32;    // causal: only key blocks <= q block

    for (int jb = 0; jb < nkb; ++jb) {
        const int kbase = jb * 32;
        __syncthreads();
#if HAVE_TDM
        // K block (32 keys x 64 d) via TDM with padded rows
        if (w == 0) {
            tdm_load_2d_bf16(lds_offset_of(Ks), Km + (rowbase + kbase) * HD,
                             HD, 32, HD, SEQ, HD, /*interval=*/4, /*pad=*/3);
        }
#else
        { // stage K block row-major
            const int r = tid >> 2;           // 0..31 key
            const int c = (tid & 3) * 16;     // 0..48 d
            const v4u* g = (const v4u*)(Km + (rowbase + kbase + r) * HD + c);
            v4u* s = (v4u*)(Ks + r * KLD + c);
            s[0] = g[0]; s[1] = g[1];
        }
#endif
        { // stage V block transposed [d][key] (TDM cannot transpose)
            const int r = tid >> 2;
            const int c = (tid & 3) * 16;
            const v4u* g = (const v4u*)(Vm + (rowbase + kbase + r) * HD + c);
            union { v4u v[2]; unsigned short s[16]; } tmp;
            tmp.v[0] = g[0]; tmp.v[1] = g[1];
#pragma unroll
            for (int j = 0; j < 16; ++j)
                Vt[(c + j) * VLD + r] = tmp.s[j];
        }
#if HAVE_TDM
        if (w == 0) __builtin_amdgcn_s_wait_tensorcnt(0);
#endif
        __syncthreads();

        // S = Q * K^T  (16 q rows x 32 keys)
        v8f sf[2];
        sf[0] = (v8f){0.f,0.f,0.f,0.f,0.f,0.f,0.f,0.f};
        sf[1] = (v8f){0.f,0.f,0.f,0.f,0.f,0.f,0.f,0.f};
#pragma unroll
        for (int ks = 0; ks < 2; ++ks) {
            FragU bk0, bk1;
            const unsigned short* p0 = Ks + l16 * KLD + ks * 32 + lhalf * 8;
            const unsigned short* p1 = Ks + (16 + l16) * KLD + ks * 32 + lhalf * 8;
            bk0.u[0] = *(const v4u*)p0; bk0.u[1] = *(const v4u*)(p0 + 16);
            bk1.u[0] = *(const v4u*)p1; bk1.u[1] = *(const v4u*)(p1 + 16);
            sf[0] = wmma_bf16(aq[ks], bk0, sf[0]);
            sf[1] = wmma_bf16(aq[ks], bk1, sf[1]);
        }

        // streaming softmax update (rows lhalf*8+r live across 16 lanes)
#pragma unroll
        for (int r = 0; r < 8; ++r) {
            const int qrow = q0 + lhalf * 8 + r;
            float s0 = sf[0][r] * QKSCALE;
            float s1 = sf[1][r] * QKSCALE;
            if (kbase + l16 > qrow)      s0 = MASK_BIAS;
            if (kbase + 16 + l16 > qrow) s1 = MASK_BIAS;
            float lmax = fmaxf(s0, s1);
#pragma unroll
            for (int off = 1; off < 16; off <<= 1)
                lmax = fmaxf(lmax, __shfl_xor(lmax, off, 32));
            const float newm = fmaxf(rm[r], lmax);
            const float corr = __expf(rm[r] - newm);
            const float p0 = __expf(s0 - newm);
            const float p1 = __expf(s1 - newm);
            float lsum = p0 + p1;
#pragma unroll
            for (int off = 1; off < 16; off <<= 1)
                lsum += __shfl_xor(lsum, off, 32);
            rl[r] = rl[r] * corr + lsum;
            rm[r] = newm;
#pragma unroll
            for (int nf = 0; nf < 4; ++nf) o[nf][r] = o[nf][r] * corr;
            // spill P to LDS for the C-layout -> A-layout shuffle
            Ps[(w * 16 + lhalf * 8 + r) * PLD + l16]      = f2bf(p0);
            Ps[(w * 16 + lhalf * 8 + r) * PLD + 16 + l16] = f2bf(p1);
        }
        __syncthreads();

        // O += P * V   (A = P 16x32, B = V^T-staged 32x64)
        FragU ap;
        const unsigned short* pp = Ps + (w * 16 + l16) * PLD + lhalf * 8;
        ap.u[0] = *(const v4u*)pp; ap.u[1] = *(const v4u*)(pp + 16);
#pragma unroll
        for (int nf = 0; nf < 4; ++nf) {
            FragU bv;
            const unsigned short* pv = Vt + (nf * 16 + l16) * VLD + lhalf * 8;
            bv.u[0] = *(const v4u*)pv; bv.u[1] = *(const v4u*)(pv + 16);
            o[nf] = wmma_bf16(ap, bv, o[nf]);
        }
    }

    // normalize and merge heads into [B,S,D] bf16
    const int b = bh / NH, h = bh % NH;
#pragma unroll
    for (int r = 0; r < 8; ++r) {
        const float inv = 1.0f / rl[r];
        const int srow = q0 + lhalf * 8 + r;
#pragma unroll
        for (int nf = 0; nf < 4; ++nf) {
            const float v = o[nf][r] * inv;
            Om[((size_t)(b * SEQ + srow)) * DMODEL + h * HD + nf * 16 + l16] = f2bf(v);
        }
    }
}

// ---------------------------------------------------------------- launch
extern "C" void kernel_launch(void* const* d_in, const int* in_sizes, int n_in,
                              void* d_out, int out_size, void* d_ws, size_t ws_size,
                              hipStream_t stream) {
    const float* hid   = (const float*)d_in[0];
    const float* attnw = (const float*)d_in[1];
    const float* attnb = (const float*)d_in[2];
    const float* projw = (const float*)d_in[3];
    const float* projb = (const float*)d_in[4];

    const size_t nX  = (size_t)BATCH * SEQ * DMODEL;     // 8388608
    const size_t nWq = (size_t)DMODEL * 3 * DMODEL;      // 3145728
    const size_t nWp = (size_t)DMODEL * DMODEL;          // 1048576

    unsigned short* Xb    = (unsigned short*)d_ws;
    unsigned short* Wqkv  = Xb + nX;
    unsigned short* Wproj = Wqkv + nWq;
    unsigned short* Qb    = Wproj + nWp;
    unsigned short* Kb    = Qb + nX;
    unsigned short* Vb    = Kb + nX;
    unsigned short* Ao    = Vb + nX;

    cast_f32_bf16<<<4096, 256, 0, stream>>>(hid,   Xb,    (int)nX);
    cast_f32_bf16<<<4096, 256, 0, stream>>>(attnw, Wqkv,  (int)nWq);
    cast_f32_bf16<<<2048, 256, 0, stream>>>(projw, Wproj, (int)nWp);

    {   // QKV projection with head-split scatter
        dim3 grid((3 * DMODEL) / 128, (BATCH * SEQ) / 128);
        gemm_bf16<1><<<grid, 256, 0, stream>>>(Xb, Wqkv, attnb,
                                               nullptr, Qb, Kb, Vb,
                                               BATCH * SEQ, 3 * DMODEL, DMODEL);
    }

    flash_attn<<<BATCH * NH * (SEQ / 64), 128, 0, stream>>>(Qb, Kb, Vb, Ao);

    {   // output projection -> f32 result
        dim3 grid(DMODEL / 128, (BATCH * SEQ) / 128);
        gemm_bf16<0><<<grid, 256, 0, stream>>>(Ao, Wproj, projb,
                                               (float*)d_out, nullptr, nullptr, nullptr,
                                               BATCH * SEQ, DMODEL, DMODEL);
    }
}
